// RQBottleneckTransformer_65927747994031
// MI455X (gfx1250) — matrix operand order repaired and checked by reading, not compiled, not run on hardware
//
#include <hip/hip_runtime.h>
#include <hip/hip_bf16.h>
#include <math.h>

// ---------------- constants ----------------
#define WIDTH 512
#define FFNW  2048
#define SEQ   1500
#define BATCH 8
#define DSQ   375           // SEQ/4
#define NTOK3 (BATCH*DSQ)   // 3000
#define NTOKB (BATCH*SEQ)   // 12000
#define CBD   32
#define NCODE 513
#define NH    8
#define HD    64
#define VT_STRIDE 1504      // SEQ padded to 16B-aligned bf16 rows
#define QK_SCALE 2.0f
#define QTILES 94           // ceil(1500/16)

typedef __attribute__((ext_vector_type(16))) __bf16 v16bf;
typedef __attribute__((ext_vector_type(8)))  float  v8f;
typedef __attribute__((ext_vector_type(4)))  unsigned int u32x4;
typedef __attribute__((ext_vector_type(8)))  int i32x8;
typedef __attribute__((ext_vector_type(4)))  int i32x4;

union FragB { int4 q[2]; v16bf v; };

__device__ __forceinline__ unsigned short f2bf(float f) {
  unsigned int u = __float_as_uint(f);
  unsigned int r = u + 0x7fffu + ((u >> 16) & 1u);
  return (unsigned short)(r >> 16);
}
__device__ __forceinline__ float geluf(float x) {
  return 0.5f * x * (1.0f + erff(x * 0.70710678118654752f));
}
__device__ __forceinline__ float redmax16(float v) {
  v = fmaxf(v, __shfl_xor(v, 1));
  v = fmaxf(v, __shfl_xor(v, 2));
  v = fmaxf(v, __shfl_xor(v, 4));
  v = fmaxf(v, __shfl_xor(v, 8));
  return v;
}
__device__ __forceinline__ float redsum16(float v) {
  v += __shfl_xor(v, 1);
  v += __shfl_xor(v, 2);
  v += __shfl_xor(v, 4);
  v += __shfl_xor(v, 8);
  return v;
}

// Issue a TDM descriptor: copy a 64-row x 32-col bf16 tile of A (row stride K
// elements) starting at (rowStart, k0) into LDS at byte offset ldsOff.
__device__ __forceinline__ void tdm_load_a_tile(const unsigned short* gbase, unsigned ldsOff,
                                                int M, int K, int rowStart, int k0) {
  size_t gaddr = (size_t)gbase + ((size_t)rowStart * K + (size_t)k0) * 2;
  u32x4 g0;
  g0[0] = 1u;                                              // count=1, user mode
  g0[1] = ldsOff;                                          // lds_addr (bytes)
  g0[2] = (unsigned)(gaddr & 0xffffffffull);               // global_addr[31:0]
  g0[3] = (unsigned)((gaddr >> 32) & 0x1ffffffull) | 0x80000000u; // addr[56:32] | type=2
  i32x8 g1;
  g1[0] = 0x00010000;                                      // data_size=1 (2 bytes), no mask
  g1[1] = (int)(((unsigned)K & 0xffffu) << 16);            // tensor_dim0[15:0]
  g1[2] = (int)((((unsigned)K >> 16) & 0xffffu) | (((unsigned)M & 0xffffu) << 16));
  g1[3] = (int)((((unsigned)M >> 16) & 0xffffu) | (32u << 16)); // tile_dim0=32
  g1[4] = 64;                                              // tile_dim1=64, tile_dim2=0
  g1[5] = (int)K;                                          // tensor_dim0_stride[31:0]
  g1[6] = 0;
  g1[7] = 0;
  i32x4 zz; zz[0] = 0; zz[1] = 0; zz[2] = 0; zz[3] = 0;
  i32x8 z8; z8[0] = 0; z8[1] = 0; z8[2] = 0; z8[3] = 0; z8[4] = 0; z8[5] = 0; z8[6] = 0; z8[7] = 0;
  __builtin_amdgcn_tensor_load_to_lds(g0, g1, zz, zz, z8, 0);
}

// ---------------- f32 -> bf16 convert ----------------
__global__ void cvt_bf_k(const float* __restrict__ in, unsigned short* __restrict__ out, int n) {
  for (int i = blockIdx.x * blockDim.x + threadIdx.x; i < n; i += gridDim.x * blockDim.x)
    out[i] = f2bf(in[i]);
}

// ---------------- downsample mean over 4 ----------------
__global__ void downsample_k(const float* __restrict__ embs, float* __restrict__ x375) {
  int row = blockIdx.x;            // 0..2999
  int c = threadIdx.x;             // 0..511
  int b = row / DSQ, t = row % DSQ;
  const float* base = embs + ((size_t)(b * SEQ + t * 4)) * WIDTH + c;
  float s = base[0] + base[WIDTH] + base[2 * WIDTH] + base[3 * WIDTH];
  x375[(size_t)row * WIDTH + c] = 0.25f * s;
}

// ---------------- layernorm over width 512 ----------------
__global__ __launch_bounds__(256) void ln_k(const float* __restrict__ x,
                                            const float* __restrict__ g,
                                            const float* __restrict__ bta,
                                            unsigned short* __restrict__ outB,
                                            float* __restrict__ outF) {
  __shared__ float red[256];
  int row = blockIdx.x, tid = threadIdx.x;
  const float* xr = x + (size_t)row * WIDTH;
  float v0 = xr[tid], v1 = xr[tid + 256];
  red[tid] = v0 + v1;
  __syncthreads();
  for (int o = 128; o > 0; o >>= 1) { if (tid < o) red[tid] += red[tid + o]; __syncthreads(); }
  float mean = red[0] * (1.0f / WIDTH);
  __syncthreads();
  float d0 = v0 - mean, d1 = v1 - mean;
  red[tid] = d0 * d0 + d1 * d1;
  __syncthreads();
  for (int o = 128; o > 0; o >>= 1) { if (tid < o) red[tid] += red[tid + o]; __syncthreads(); }
  float rstd = rsqrtf(red[0] * (1.0f / WIDTH) + 1e-5f);
  float y0 = d0 * rstd * g[tid] + bta[tid];
  float y1 = d1 * rstd * g[tid + 256] + bta[tid + 256];
  if (outB) { outB[(size_t)row * WIDTH + tid] = f2bf(y0); outB[(size_t)row * WIDTH + tid + 256] = f2bf(y1); }
  if (outF) { outF[(size_t)row * WIDTH + tid] = y0; outF[(size_t)row * WIDTH + tid + 256] = y1; }
}

// =====================================================================
// WMMA bf16 GEMM (direct global loads): out = act(A*W^T + bias [+resid])
// =====================================================================
__global__ __launch_bounds__(256) void gemm_bf16_wmma(
    const unsigned short* __restrict__ A, const unsigned short* __restrict__ W,
    const float* __restrict__ bias, const float* __restrict__ resid,
    float* __restrict__ outF, unsigned short* __restrict__ outB,
    int M, int N, int K, int gelu) {
  int wave = threadIdx.x >> 5, lane = threadIdx.x & 31;
  int hf = lane >> 4, l = lane & 15;
  int m0 = blockIdx.x * 64 + (wave & 1) * 32;
  int n0 = blockIdx.y * 128 + (wave >> 1) * 32;
  v8f acc[2][2];
#pragma unroll
  for (int i = 0; i < 2; i++)
#pragma unroll
    for (int j = 0; j < 2; j++)
#pragma unroll
      for (int e = 0; e < 8; e++) acc[i][j][e] = 0.0f;

  for (int k0 = 0; k0 < K; k0 += 32) {
    FragB a[2], b[2];
#pragma unroll
    for (int mt = 0; mt < 2; mt++) {
      int row = m0 + mt * 16 + l; if (row >= M) row = M - 1;
      const unsigned short* p = A + (size_t)row * K + k0 + hf * 16;
      a[mt].q[0] = *reinterpret_cast<const int4*>(p);
      a[mt].q[1] = *reinterpret_cast<const int4*>(p + 8);
    }
#pragma unroll
    for (int nt = 0; nt < 2; nt++) {
      int col = n0 + nt * 16 + l;
      const unsigned short* p = W + (size_t)col * K + k0 + hf * 16;
      b[nt].q[0] = *reinterpret_cast<const int4*>(p);
      b[nt].q[1] = *reinterpret_cast<const int4*>(p + 8);
    }
    if (k0 + 32 < K) {
      int prow = m0 + l; if (prow >= M) prow = M - 1;
      __builtin_prefetch(A + (size_t)prow * K + k0 + 32, 0, 1);
      __builtin_prefetch(W + (size_t)(n0 + l) * K + k0 + 32, 0, 1);
    }
#pragma unroll
    for (int mt = 0; mt < 2; mt++)
#pragma unroll
      for (int nt = 0; nt < 2; nt++)
        acc[mt][nt] = __builtin_amdgcn_wmma_f32_16x16x32_bf16(
            false, a[mt].v, false, b[nt].v, (short)0, acc[mt][nt], false, false);
  }
#pragma unroll
  for (int mt = 0; mt < 2; mt++) {
#pragma unroll
    for (int nt = 0; nt < 2; nt++) {
      int col = n0 + nt * 16 + l;
      float bb = bias ? bias[col] : 0.0f;
#pragma unroll
      for (int r = 0; r < 8; r++) {
        int row = m0 + mt * 16 + hf * 8 + r;
        if (row < M) {
          float v = acc[mt][nt][r] + bb;
          if (resid) v += resid[(size_t)row * N + col];
          if (gelu) v = geluf(v);
          if (outF) outF[(size_t)row * N + col] = v;
          if (outB) outB[(size_t)row * N + col] = f2bf(v);
        }
      }
    }
  }
}

// =====================================================================
// WMMA bf16 GEMM with TDM-staged A tiles (double-buffered LDS).
// One wave per block issues tensor_load_to_lds for the 64x32 A tile of
// the NEXT K-step while all 8 waves compute from the current tile.
// =====================================================================
__global__ __launch_bounds__(256) void gemm_bf16_wmma_tdm(
    const unsigned short* __restrict__ A, const unsigned short* __restrict__ W,
    const float* __restrict__ bias, const float* __restrict__ resid,
    float* __restrict__ outF, unsigned short* __restrict__ outB,
    int M, int N, int K, int gelu) {
  __shared__ unsigned short abuf[2][64 * 32];
  int wave = threadIdx.x >> 5, lane = threadIdx.x & 31;
  int hf = lane >> 4, l = lane & 15;
  int m0loc = (wave & 1) * 32;
  int rowStart = blockIdx.x * 64;
  int n0 = blockIdx.y * 128 + (wave >> 1) * 32;
  unsigned ldsOff0 = (unsigned)(size_t)(void*)&abuf[0][0];
  unsigned ldsOff1 = (unsigned)(size_t)(void*)&abuf[1][0];

  v8f acc[2][2];
#pragma unroll
  for (int i = 0; i < 2; i++)
#pragma unroll
    for (int j = 0; j < 2; j++)
#pragma unroll
      for (int e = 0; e < 8; e++) acc[i][j][e] = 0.0f;

  int nsteps = K >> 5;
  if (wave == 0) tdm_load_a_tile(A, ldsOff0, M, K, rowStart, 0);

  int cur = 0;
  for (int s = 0; s < nsteps; s++) {
    int k0 = s << 5;
    if (wave == 0) __builtin_amdgcn_s_wait_tensorcnt(0);
    __syncthreads();
    if (wave == 0 && s + 1 < nsteps)
      tdm_load_a_tile(A, cur ? ldsOff0 : ldsOff1, M, K, rowStart, k0 + 32);

    FragB a[2], b[2];
#pragma unroll
    for (int mt = 0; mt < 2; mt++) {
      const unsigned short* p = &abuf[cur][(m0loc + mt * 16 + l) * 32 + hf * 16];
      a[mt].q[0] = *reinterpret_cast<const int4*>(p);
      a[mt].q[1] = *reinterpret_cast<const int4*>(p + 8);
    }
#pragma unroll
    for (int nt = 0; nt < 2; nt++) {
      int col = n0 + nt * 16 + l;
      const unsigned short* p = W + (size_t)col * K + k0 + hf * 16;
      b[nt].q[0] = *reinterpret_cast<const int4*>(p);
      b[nt].q[1] = *reinterpret_cast<const int4*>(p + 8);
    }
    if (k0 + 32 < K)
      __builtin_prefetch(W + (size_t)(n0 + l) * K + k0 + 32, 0, 1);
#pragma unroll
    for (int mt = 0; mt < 2; mt++)
#pragma unroll
      for (int nt = 0; nt < 2; nt++)
        acc[mt][nt] = __builtin_amdgcn_wmma_f32_16x16x32_bf16(
            false, a[mt].v, false, b[nt].v, (short)0, acc[mt][nt], false, false);
    cur ^= 1;
  }
#pragma unroll
  for (int mt = 0; mt < 2; mt++) {
#pragma unroll
    for (int nt = 0; nt < 2; nt++) {
      int col = n0 + nt * 16 + l;
      float bb = bias ? bias[col] : 0.0f;
#pragma unroll
      for (int r = 0; r < 8; r++) {
        int row = rowStart + m0loc + mt * 16 + hf * 8 + r;
        if (row < M) {
          float v = acc[mt][nt][r] + bb;
          if (resid) v += resid[(size_t)row * N + col];
          if (gelu) v = geluf(v);
          if (outF) outF[(size_t)row * N + col] = v;
          if (outB) outB[(size_t)row * N + col] = f2bf(v);
        }
      }
    }
  }
}

// ---------------- proj_in: z[3000,32] = x375 @ Wpi^T + b ----------------
__global__ void proj_in_k(const float* __restrict__ x, const float* __restrict__ w,
                          const float* __restrict__ b, float* __restrict__ z) {
  int row = blockIdx.x, c = threadIdx.x;  // grid 3000, block 32
  float s = b[c];
  const float* xr = x + (size_t)row * WIDTH;
  const float* wr = w + (size_t)c * WIDTH;
  for (int k = 0; k < WIDTH; k++) s += xr[k] * wr[k];
  z[(size_t)row * CBD + c] = s;
}

// ---------------- VQ nearest code + commit loss ----------------
__global__ void vq_k(const float* __restrict__ z, const float* __restrict__ cb,
                     float* __restrict__ zq, unsigned short* __restrict__ zq_bf,
                     float* __restrict__ out_idx, float* __restrict__ out_loss) {
  for (int row = blockIdx.x * blockDim.x + threadIdx.x; row < NTOK3; row += gridDim.x * blockDim.x) {
    float zv[CBD];
#pragma unroll
    for (int k = 0; k < CBD; k++) zv[k] = z[(size_t)row * CBD + k];
    float best = 3.0e38f; int bi = 0;
    for (int c = 0; c < NCODE; c++) {
      float d = 0.0f;
      const float* cr = cb + (size_t)c * CBD;
#pragma unroll
      for (int k = 0; k < CBD; k++) { float df = zv[k] - cr[k]; d += df * df; }
      if (d < best) { best = d; bi = c; }
    }
    const float* cr = cb + (size_t)bi * CBD;
#pragma unroll
    for (int k = 0; k < CBD; k++) { zq[(size_t)row * CBD + k] = cr[k]; zq_bf[(size_t)row * CBD + k] = f2bf(cr[k]); }
    out_idx[row] = (float)bi;
    atomicAdd(out_loss, best * (1.0f / (NTOK3 * CBD)));
  }
}

__global__ void zero_loss_k(float* p) { if (threadIdx.x == 0) *p = 0.0f; }

// ---------------- mask code vector ----------------
__global__ void mask_code_k(const float* __restrict__ cb, const float* __restrict__ wpo,
                            const float* __restrict__ bpo, float* __restrict__ mc) {
  int c = threadIdx.x + blockIdx.x * blockDim.x;
  if (c < WIDTH) {
    float s = bpo[c];
    const float* code = cb + (size_t)(NCODE - 1) * CBD;
#pragma unroll
    for (int k = 0; k < CBD; k++) s += code[k] * wpo[(size_t)c * CBD + k];
    mc[c] = s;
  }
}

// ---------------- upsample + mask + pos_emb ----------------
__global__ void upsample_k(const float* __restrict__ qz, const unsigned char* __restrict__ mask,
                           const float* __restrict__ mc, const float* __restrict__ pos,
                           float* __restrict__ xb) {
  int gt = blockIdx.x, c = threadIdx.x;   // grid 12000, block 512
  int b = gt / SEQ, t = gt % SEQ;
  float v = mask[gt] ? qz[(size_t)(b * DSQ + t / 4) * WIDTH + c] : mc[c];
  xb[(size_t)gt * WIDTH + c] = v + pos[(size_t)t * WIDTH + c];
}

// ---------------- RoPE + QK scale + bf16 ----------------
__global__ void rope_k(const float* __restrict__ qf, const float* __restrict__ kf,
                       unsigned short* __restrict__ qb, unsigned short* __restrict__ kb) {
  const int n = BATCH * SEQ * NH * (HD / 2);
  for (int idx = blockIdx.x * blockDim.x + threadIdx.x; idx < n; idx += gridDim.x * blockDim.x) {
    int i = idx % (HD / 2);
    int h = (idx / (HD / 2)) % NH;
    int t = (idx / (HD / 2 * NH)) % SEQ;
    int b = idx / (HD / 2 * NH * SEQ);
    float inv = __expf(-logf(10000.0f) * (2.0f * i) / (float)HD);
    float ang = (float)t * inv;
    float cs = __cosf(ang), sn = __sinf(ang);
    size_t base = ((size_t)(b * SEQ + t)) * WIDTH + h * HD + 2 * i;
    float x1 = qf[base], x2 = qf[base + 1];
    qb[base]     = f2bf((x1 * cs - x2 * sn) * QK_SCALE);
    qb[base + 1] = f2bf((x1 * sn + x2 * cs) * QK_SCALE);
    x1 = kf[base]; x2 = kf[base + 1];
    kb[base]     = f2bf(x1 * cs - x2 * sn);
    kb[base + 1] = f2bf(x1 * sn + x2 * cs);
  }
}

// ---------------- V transpose to [b][h][d][t] bf16 (padded stride) ----------------
__global__ void vtrans_k(const float* __restrict__ vf, unsigned short* __restrict__ vt) {
  const int n = BATCH * NH * HD * SEQ;
  for (int idx = blockIdx.x * blockDim.x + threadIdx.x; idx < n; idx += gridDim.x * blockDim.x) {
    int t = idx % SEQ;
    int d = (idx / SEQ) % HD;
    int h = (idx / (SEQ * HD)) % NH;
    int b = idx / (SEQ * HD * NH);
    vt[((size_t)((b * NH + h) * HD + d)) * VT_STRIDE + t] =
        f2bf(vf[((size_t)(b * SEQ + t)) * WIDTH + h * HD + d]);
  }
}

// ---------------- flash attention (bf16 WMMA, f32 online softmax) ----------------
__global__ __launch_bounds__(256) void attn_k(const unsigned short* __restrict__ qb,
                                              const unsigned short* __restrict__ kb,
                                              const unsigned short* __restrict__ vt,
                                              unsigned short* __restrict__ ob) {
  __shared__ unsigned short pbuf[8 * 16 * 32];
  int wave = threadIdx.x >> 5, lane = threadIdx.x & 31;
  int hf = lane >> 4, l = lane & 15;
  int task = blockIdx.x * 8 + wave;
  if (task >= BATCH * NH * QTILES) return;
  int qt = task % QTILES;
  int h = (task / QTILES) % NH;
  int b = task / (QTILES * NH);
  int q0 = qt * 16;

  FragB aq[2];
  {
    int t = q0 + l; if (t > SEQ - 1) t = SEQ - 1;
    const unsigned short* p = qb + ((size_t)(b * SEQ + t)) * WIDTH + h * HD;
#pragma unroll
    for (int kd = 0; kd < 2; kd++) {
      const unsigned short* pp = p + kd * 32 + hf * 16;
      aq[kd].q[0] = *reinterpret_cast<const int4*>(pp);
      aq[kd].q[1] = *reinterpret_cast<const int4*>(pp + 8);
    }
  }

  v8f acc[4];
#pragma unroll
  for (int d = 0; d < 4; d++)
#pragma unroll
    for (int e = 0; e < 8; e++) acc[d][e] = 0.0f;
  float mlo[8], mhi[8], llo[8], lhi[8];
#pragma unroll
  for (int r = 0; r < 8; r++) { mlo[r] = -3.0e38f; mhi[r] = -3.0e38f; llo[r] = 0.0f; lhi[r] = 0.0f; }

  unsigned short* pw = pbuf + wave * 512;

  for (int kb0 = 0; kb0 < SEQ; kb0 += 32) {
    v8f S[2];
#pragma unroll
    for (int j = 0; j < 2; j++) {
      int key = kb0 + j * 16 + l;
      int kc = key > SEQ - 1 ? SEQ - 1 : key;
      const unsigned short* p = kb + ((size_t)(b * SEQ + kc)) * WIDTH + h * HD;
      FragB bk0, bk1;
      bk0.q[0] = *reinterpret_cast<const int4*>(p + hf * 16);
      bk0.q[1] = *reinterpret_cast<const int4*>(p + hf * 16 + 8);
      bk1.q[0] = *reinterpret_cast<const int4*>(p + 32 + hf * 16);
      bk1.q[1] = *reinterpret_cast<const int4*>(p + 32 + hf * 16 + 8);
      v8f z;
#pragma unroll
      for (int e = 0; e < 8; e++) z[e] = 0.0f;
      z = __builtin_amdgcn_wmma_f32_16x16x32_bf16(false, aq[0].v, false, bk0.v, (short)0, z, false, false);
      z = __builtin_amdgcn_wmma_f32_16x16x32_bf16(false, aq[1].v, false, bk1.v, (short)0, z, false, false);
      if (key > SEQ - 1) {
#pragma unroll
        for (int e = 0; e < 8; e++) z[e] = -3.0e38f;
      }
      S[j] = z;
    }
    // per-row max (rows r / r+8 live on lane halves)
    float nmlo[8], nmhi[8], sclo[8], schi[8];
#pragma unroll
    for (int r = 0; r < 8; r++) {
      float v = fmaxf(S[0][r], S[1][r]);
      v = redmax16(v);
      float lo = __shfl(v, 0), hi = __shfl(v, 16);
      nmlo[r] = fmaxf(mlo[r], lo);
      nmhi[r] = fmaxf(mhi[r], hi);
      sclo[r] = __expf(mlo[r] - nmlo[r]);
      schi[r] = __expf(mhi[r] - nmhi[r]);
      mlo[r] = nmlo[r]; mhi[r] = nmhi[r];
    }
    // probabilities -> LDS, row sums
#pragma unroll
    for (int r = 0; r < 8; r++) {
      float mrow = hf ? nmhi[r] : nmlo[r];
      float p0 = __expf(S[0][r] - mrow);
      float p1 = __expf(S[1][r] - mrow);
      int rr = r + 8 * hf;
      pw[rr * 32 + l] = f2bf(p0);
      pw[rr * 32 + 16 + l] = f2bf(p1);
      float sv = redsum16(p0 + p1);
      float slo = __shfl(sv, 0), shi = __shfl(sv, 16);
      llo[r] = llo[r] * sclo[r] + slo;
      lhi[r] = lhi[r] * schi[r] + shi;
    }
    // rescale accumulators
#pragma unroll
    for (int d = 0; d < 4; d++)
#pragma unroll
      for (int r = 0; r < 8; r++) acc[d][r] *= (hf ? schi[r] : sclo[r]);
    // reload P as A fragment (16x32)
    FragB ap;
    {
      const unsigned short* pp = pw + l * 32 + hf * 16;
      ap.q[0] = *reinterpret_cast<const int4*>(pp);
      ap.q[1] = *reinterpret_cast<const int4*>(pp + 8);
    }
    // P @ V
#pragma unroll
    for (int d0 = 0; d0 < 4; d0++) {
      const unsigned short* pv = vt + ((size_t)((b * NH + h) * HD + d0 * 16 + l)) * VT_STRIDE + kb0 + hf * 16;
      FragB bv;
      bv.q[0] = *reinterpret_cast<const int4*>(pv);
      bv.q[1] = *reinterpret_cast<const int4*>(pv + 8);
      acc[d0] = __builtin_amdgcn_wmma_f32_16x16x32_bf16(false, ap.v, false, bv.v, (short)0, acc[d0], false, false);
    }
  }
  // write O (bf16)
#pragma unroll
  for (int d0 = 0; d0 < 4; d0++)
#pragma unroll
    for (int r = 0; r < 8; r++) {
      int t = q0 + r + 8 * hf;
      if (t < SEQ) {
        float denom = hf ? lhi[r] : llo[r];
        float v = acc[d0][r] / denom;
        ob[((size_t)(b * SEQ + t)) * WIDTH + h * HD + d0 * 16 + l] = f2bf(v);
      }
    }
}

// ======================= host launcher =======================
extern "C" void kernel_launch(void* const* d_in, const int* in_sizes, int n_in,
                              void* d_out, int out_size, void* d_ws, size_t ws_size,
                              hipStream_t stream) {
  const float* embs       = (const float*)d_in[0];
  const unsigned char* mask = (const unsigned char*)d_in[1];
  const float* mlp_w1 = (const float*)d_in[2];  const float* mlp_b1 = (const float*)d_in[3];
  const float* mlp_w2 = (const float*)d_in[4];  const float* mlp_b2 = (const float*)d_in[5];
  const float* mlp_ln_g = (const float*)d_in[6];const float* mlp_ln_b = (const float*)d_in[7];
  const float* proj_in_w = (const float*)d_in[8]; const float* proj_in_b = (const float*)d_in[9];
  const float* codebook = (const float*)d_in[10];
  const float* proj_out_w = (const float*)d_in[11]; const float* proj_out_b = (const float*)d_in[12];
  const float* pos_emb = (const float*)d_in[13];
  const float* attn_ln_g = (const float*)d_in[14]; const float* attn_ln_b = (const float*)d_in[15];
  const float* wq = (const float*)d_in[16]; const float* bq = (const float*)d_in[17];
  const float* wk = (const float*)d_in[18];
  const float* wv = (const float*)d_in[19]; const float* bv = (const float*)d_in[20];
  const float* wo = (const float*)d_in[21]; const float* bo = (const float*)d_in[22];
  const float* ffn_ln_g = (const float*)d_in[23]; const float* ffn_ln_b = (const float*)d_in[24];
  const float* ffn_w1 = (const float*)d_in[25]; const float* ffn_b1 = (const float*)d_in[26];
  const float* ffn_w2 = (const float*)d_in[27]; const float* ffn_b2 = (const float*)d_in[28];
  const float* ln_post_g = (const float*)d_in[29]; const float* ln_post_b = (const float*)d_in[30];

  float* out_x   = (float*)d_out;                                  // [8,1500,512]
  float* out_idx = out_x + (size_t)NTOKB * WIDTH;                  // [3000]
  float* out_loss = out_idx + NTOK3;                               // [1]

  // workspace carve-up
  char* w = (char*)d_ws; size_t off = 0;
  auto alloc = [&](size_t bytes) { void* p = w + off; off += (bytes + 255) & ~(size_t)255; return p; };
  float* x375  = (float*)alloc((size_t)NTOK3 * WIDTH * 4);
  unsigned short* xn_bf = (unsigned short*)alloc((size_t)NTOKB * WIDTH * 2);
  unsigned short* h_bf  = (unsigned short*)alloc((size_t)NTOKB * FFNW * 2);
  unsigned short* w_mlp1 = (unsigned short*)alloc((size_t)FFNW * WIDTH * 2);
  unsigned short* w_mlp2 = (unsigned short*)alloc((size_t)WIDTH * FFNW * 2);
  unsigned short* w_q = (unsigned short*)alloc((size_t)WIDTH * WIDTH * 2);
  unsigned short* w_k = (unsigned short*)alloc((size_t)WIDTH * WIDTH * 2);
  unsigned short* w_v = (unsigned short*)alloc((size_t)WIDTH * WIDTH * 2);
  unsigned short* w_o = (unsigned short*)alloc((size_t)WIDTH * WIDTH * 2);
  unsigned short* w_f1 = (unsigned short*)alloc((size_t)FFNW * WIDTH * 2);
  unsigned short* w_f2 = (unsigned short*)alloc((size_t)WIDTH * FFNW * 2);
  unsigned short* w_po = (unsigned short*)alloc((size_t)WIDTH * CBD * 2);
  float* z   = (float*)alloc((size_t)NTOK3 * CBD * 4);
  float* zq  = (float*)alloc((size_t)NTOK3 * CBD * 4);
  unsigned short* zq_bf = (unsigned short*)alloc((size_t)NTOK3 * CBD * 2);
  float* qz  = (float*)alloc((size_t)NTOK3 * WIDTH * 4);
  float* mc  = (float*)alloc((size_t)WIDTH * 4);
  float* x_big = (float*)alloc((size_t)NTOKB * WIDTH * 4);
  float* qf = (float*)alloc((size_t)NTOKB * WIDTH * 4);
  float* kf = (float*)alloc((size_t)NTOKB * WIDTH * 4);
  float* vf = (float*)alloc((size_t)NTOKB * WIDTH * 4);
  unsigned short* q_bf = (unsigned short*)alloc((size_t)NTOKB * WIDTH * 2);
  unsigned short* k_bf = (unsigned short*)alloc((size_t)NTOKB * WIDTH * 2);
  unsigned short* v_t  = (unsigned short*)alloc((size_t)BATCH * NH * HD * VT_STRIDE * 2);
  unsigned short* o_bf = (unsigned short*)alloc((size_t)NTOKB * WIDTH * 2);
  (void)ws_size; (void)n_in; (void)in_sizes; (void)out_size;

  auto gemm_tdm = [&](const unsigned short* A, const unsigned short* W_, const float* bias,
                      const float* resid, float* oF, unsigned short* oB, int M, int N, int K, int gelu) {
    dim3 grid((M + 63) / 64, N / 128);
    gemm_bf16_wmma_tdm<<<grid, 256, 0, stream>>>(A, W_, bias, resid, oF, oB, M, N, K, gelu);
  };
  auto gemm_direct = [&](const unsigned short* A, const unsigned short* W_, const float* bias,
                         const float* resid, float* oF, unsigned short* oB, int M, int N, int K, int gelu) {
    dim3 grid((M + 63) / 64, N / 128);
    gemm_bf16_wmma<<<grid, 256, 0, stream>>>(A, W_, bias, resid, oF, oB, M, N, K, gelu);
  };
  auto cvt = [&](const float* src, unsigned short* dst, int n) {
    cvt_bf_k<<<512, 256, 0, stream>>>(src, dst, n);
  };

  zero_loss_k<<<1, 32, 0, stream>>>(out_loss);

  // weight conversions
  cvt(mlp_w1, w_mlp1, FFNW * WIDTH);
  cvt(mlp_w2, w_mlp2, WIDTH * FFNW);
  cvt(wq, w_q, WIDTH * WIDTH);
  cvt(wk, w_k, WIDTH * WIDTH);
  cvt(wv, w_v, WIDTH * WIDTH);
  cvt(wo, w_o, WIDTH * WIDTH);
  cvt(ffn_w1, w_f1, FFNW * WIDTH);
  cvt(ffn_w2, w_f2, WIDTH * FFNW);
  cvt(proj_out_w, w_po, WIDTH * CBD);

  // downsample + MLP block
  downsample_k<<<NTOK3, WIDTH, 0, stream>>>(embs, x375);
  ln_k<<<NTOK3, 256, 0, stream>>>(x375, mlp_ln_g, mlp_ln_b, xn_bf, nullptr);
  gemm_tdm(xn_bf, w_mlp1, mlp_b1, nullptr, nullptr, h_bf, NTOK3, FFNW, WIDTH, 1);
  gemm_tdm(h_bf, w_mlp2, mlp_b2, x375, x375, nullptr, NTOK3, WIDTH, FFNW, 0);

  // VQ bottleneck
  proj_in_k<<<NTOK3, CBD, 0, stream>>>(x375, proj_in_w, proj_in_b, z);
  vq_k<<<128, 256, 0, stream>>>(z, codebook, zq, zq_bf, out_idx, out_loss);
  mask_code_k<<<2, 256, 0, stream>>>(codebook, proj_out_w, proj_out_b, mc);
  gemm_direct(zq_bf, w_po, proj_out_b, nullptr, qz, nullptr, NTOK3, WIDTH, CBD, 0);
  upsample_k<<<NTOKB, WIDTH, 0, stream>>>(qz, mask, mc, pos_emb, x_big);

  // attention block
  ln_k<<<NTOKB, 256, 0, stream>>>(x_big, attn_ln_g, attn_ln_b, xn_bf, nullptr);
  gemm_tdm(xn_bf, w_q, bq, nullptr, qf, nullptr, NTOKB, WIDTH, WIDTH, 0);
  gemm_tdm(xn_bf, w_k, nullptr, nullptr, kf, nullptr, NTOKB, WIDTH, WIDTH, 0);
  gemm_tdm(xn_bf, w_v, bv, nullptr, vf, nullptr, NTOKB, WIDTH, WIDTH, 0);
  rope_k<<<1024, 256, 0, stream>>>(qf, kf, q_bf, k_bf);
  vtrans_k<<<1024, 256, 0, stream>>>(vf, v_t);
  {
    int tasks = BATCH * NH * QTILES;
    attn_k<<<(tasks + 7) / 8, 256, 0, stream>>>(q_bf, k_bf, v_t, o_bf);
  }
  gemm_tdm(o_bf, w_o, bo, x_big, x_big, nullptr, NTOKB, WIDTH, WIDTH, 0);

  // FFN block
  ln_k<<<NTOKB, 256, 0, stream>>>(x_big, ffn_ln_g, ffn_ln_b, xn_bf, nullptr);
  gemm_tdm(xn_bf, w_f1, ffn_b1, nullptr, nullptr, h_bf, NTOKB, FFNW, WIDTH, 1);
  gemm_tdm(h_bf, w_f2, ffn_b2, x_big, x_big, nullptr, NTOKB, WIDTH, FFNW, 0);

  // final layernorm -> output
  ln_k<<<NTOKB, 256, 0, stream>>>(x_big, ln_post_g, ln_post_b, nullptr, out_x);
}
